// DiffSchNet_27839978013476
// MI455X (gfx1250) — matrix operand order repaired and checked by reading, not compiled, not run on hardware
//
#include <hip/hip_runtime.h>
#include <hip/hip_bf16.h>

typedef _Float16 f16;
typedef __attribute__((ext_vector_type(16))) _Float16 v16h;
typedef __attribute__((ext_vector_type(8)))  _Float16 v8h;
typedef __attribute__((ext_vector_type(8)))  float    v8f;

#define NBATCH 32
#define NELEC 128
#define NNUC 32
#define EMB 128
#define KER 128
#define INW 224        // 7 * 32 basis features
#define HID_REAL 169
#define HID_PAD 192
#define KT1 7          // 224/32 K-tiles, GEMM1
#define NT1 12         // 192/16 N-tiles, GEMM1
#define KT2 6          // 192/32 K-tiles, GEMM2
#define NT2 8          // 128/16 N-tiles, GEMM2
#define KTE 4          // 128/32 K-tiles, elec GEMMs
#define NTE 8          // 128/16 N-tiles, elec GEMMs
#define HPITCH 200     // padded LDS hidden row pitch (halfs)
#define E_SAME 8064
#define E_ANTI 8192
#define E_NE   4096
#define W1_HALFS ((size_t)KT1 * NT1 * 512)   // 43008 halfs = 86016 B
#define W1_LDS_BYTES (KT1 * NT1 * 512 * 2)   // dynamic LDS for staged W1

static __device__ __forceinline__ float silu_f(float x) {
    // x * sigmoid(x), fast hardware reciprocal (v_rcp_f32), no div fixup
    return x * __builtin_amdgcn_rcpf(1.f + __expf(-x));
}

// ---------------------------------------------------------------------------
// Convert an fp32 weight matrix (K x N, row-major) into f16 B-fragment order:
// dst[((kt*NT + nt)*32 + lane)*16 + e] = W[kt*32 + 8*(lane>=16) + e + 8*(e>=8)][nt*16 + (lane&15)]
// Zero-pads rows >= realK and cols >= realN.
// ---------------------------------------------------------------------------
__global__ void swizzle_w(const float* __restrict__ src, f16* __restrict__ dst,
                          int NT, int realK, int realN) {
    int t    = blockIdx.x * blockDim.x + threadIdx.x;
    int e    = t & 15;
    int lane = (t >> 4) & 31;
    int tile = t >> 9;
    int kt = tile / NT, nt = tile % NT;
    int row = kt * 32 + ((lane >> 4) << 3) + e + ((e >> 3) << 3);
    int col = nt * 16 + (lane & 15);
    float v = (row < realK && col < realN) ? src[row * realN + col] : 0.f;
    dst[t] = (f16)v;
}

// ---------------------------------------------------------------------------
// Geometric edge features (computed once, shared by all 3 layers).
// ---------------------------------------------------------------------------
__global__ void feat_kernel(const float* __restrict__ rs, const float* __restrict__ nucxyz,
                            const int* __restrict__ send, const int* __restrict__ recv,
                            f16* __restrict__ feats, int E, int sendIsNuc) {
    int q    = threadIdx.x & 31;
    int edge = blockIdx.x * 8 + (threadIdx.x >> 5);
    int b    = blockIdx.y;
    if (edge >= E) return;
    int s = send[edge], r = recv[edge];
    const float* rp = rs + ((size_t)b * NELEC + r) * 3;
    float sx, sy, sz;
    if (sendIsNuc) { sx = nucxyz[s*3+0]; sy = nucxyz[s*3+1]; sz = nucxyz[s*3+2]; }
    else { const float* sp = rs + ((size_t)b * NELEC + s) * 3; sx = sp[0]; sy = sp[1]; sz = sp[2]; }
    float dx = rp[0] - sx, dy = rp[1] - sy, dz = rp[2] - sz;
    float dn = sqrtf(dx*dx + dy*dy + dz*dz);
    const float delta = 1.f / 64.f;
    float qs = delta + q * ((1.f - 2.f * delta) / 31.f);
    float mu = 10.f * qs * qs;
    float sg = (1.f + 10.f * qs) / 7.f;
    float inv_s2 = 1.f / (sg * sg);
    f16* out = feats + ((size_t)b * E + edge) * INW + q;
    float c[3] = {dx, dy, dz};
#pragma unroll
    for (int i = 0; i < 3; ++i) {
        float dp = c[i] > 0.f ?  c[i] : 0.f;
        float dm = c[i] < 0.f ? -c[i] : 0.f;
        float fp = dp*dp*__expf(-dp) * __expf(-(dp-mu)*(dp-mu)*inv_s2);
        float fm = dm*dm*__expf(-dm) * __expf(-(dm-mu)*(dm-mu)*inv_s2);
        out[(2*i  ) * 32] = (f16)fp;
        out[(2*i+1) * 32] = (f16)fm;
    }
    float fn = dn*dn*__expf(-dn) * __expf(-(dn-mu)*(dn-mu)*inv_s2);
    out[6 * 32] = (f16)fn;
}

__global__ void init_elec(float* __restrict__ elec, const float* __restrict__ X) {
    int t = blockIdx.x * blockDim.x + threadIdx.x;
    elec[t] = X[t & (EMB - 1)];
}

__global__ void zero_f32(float* __restrict__ p, int n) {
    int t = blockIdx.x * blockDim.x + threadIdx.x;
    if (t < n) p[t] = 0.f;
}

// B-fragment load from pre-swizzled weights (32B contiguous per lane)
#define LDB(base, NT, kt, nt) \
    (*(const v16h*)((base) + (((size_t)(kt) * (NT) + (nt)) * 32 + lane) * 16))

// ---------------------------------------------------------------------------
// Core kernel: per-edge MLP (224 -> silu(169) -> 128) via WMMA, fused with
// the elementwise hx[sender] (or nuc[sender]) multiply and the atomic
// segment-sum scatter into z[receiver].
//
// W1 fragments (86KB, the bulk of B traffic) are cooperatively staged into
// dynamic LDS once per workgroup: the GEMM1 hot loop then runs entirely on
// ds_load_b128 B reads (short latency) feeding 4 independent WMMA chains.
// 1 wave = 32 edges (two A-tiles sharing each B fragment); 4 waves/WG.
// Total LDS = 86016 (dyn W1) + 51200 (hidden) = 137216 B -> 2 WGs per WGP.
// ---------------------------------------------------------------------------
__global__ __launch_bounds__(128, 1) void edge_mlp(
    const f16* __restrict__ feats, const f16* __restrict__ w1sw,
    const float* __restrict__ b1, const f16* __restrict__ w2sw,
    const float* __restrict__ mult, int multPerBatch,
    const int* __restrict__ send, const int* __restrict__ recv,
    float* __restrict__ z, int E)
{
    extern __shared__ __align__(16) char smem[];
    f16* ldsW1 = (f16*)smem;                       // [KT1*NT1*32][16] fragments
    __shared__ alignas(16) f16 hid[4][2][16][HPITCH];

    const int lane = threadIdx.x & 31;
    const int wv   = threadIdx.x >> 5;
    const int ln   = lane & 15;
    const int hi   = lane >> 4;      // 0 for lanes 0-15, 1 for lanes 16-31
    const int b    = blockIdx.y;
    const int edgeBase = (blockIdx.x * 4 + wv) * 32;   // 32 edges per wave

    // ---- cooperative stage of W1 fragments into LDS (2688 x 32B chunks) ----
    for (int i = threadIdx.x; i < KT1 * NT1 * 32; i += 128)
        *(v16h*)(ldsW1 + (size_t)i * 16) = *(const v16h*)(w1sw + (size_t)i * 16);

    // ---- A1 fragments: 2 tiles x 16 edges x 224 features (f16, registers,
    //      non-temporal: single-use stream, keep weights cached) ----
    v16h a1[2][KT1];
#pragma unroll
    for (int t = 0; t < 2; ++t) {
        const f16* arow = feats + ((size_t)b * E + edgeBase + t * 16 + ln) * INW;
#pragma unroll
        for (int kt = 0; kt < KT1; ++kt) {
            v8h c0 = __builtin_nontemporal_load((const v8h*)(arow + kt * 32 + hi * 8));
            v8h c1 = __builtin_nontemporal_load((const v8h*)(arow + kt * 32 + 16 + hi * 8));
#pragma unroll
            for (int j = 0; j < 8; ++j) { a1[t][kt][j] = c0[j]; a1[t][kt][8 + j] = c1[j]; }
        }
    }
    __syncthreads();

    // ---- GEMM1 (B from LDS) + bias + silu, hidden (f16) to per-wave LDS ----
    for (int np = 0; np < NT1; np += 2) {
        int col0 = np * 16 + ln;
        int col1 = col0 + 16;
        float bias0 = (col0 < HID_REAL) ? b1[col0] : 0.f;
        float bias1 = (col1 < HID_REAL) ? b1[col1] : 0.f;
        v8f acc00, acc01, acc10, acc11;
#pragma unroll
        for (int r2 = 0; r2 < 8; ++r2) {
            acc00[r2] = bias0; acc10[r2] = bias0;
            acc01[r2] = bias1; acc11[r2] = bias1;
        }
#pragma unroll
        for (int kt = 0; kt < KT1; ++kt) {
            v16h bf0 = LDB(ldsW1, NT1, kt, np);
            v16h bf1 = LDB(ldsW1, NT1, kt, np + 1);
            acc00 = __builtin_amdgcn_wmma_f32_16x16x32_f16(false, a1[0][kt], false, bf0, (short)0, acc00, false, false);
            acc10 = __builtin_amdgcn_wmma_f32_16x16x32_f16(false, a1[1][kt], false, bf0, (short)0, acc10, false, false);
            acc01 = __builtin_amdgcn_wmma_f32_16x16x32_f16(false, a1[0][kt], false, bf1, (short)0, acc01, false, false);
            acc11 = __builtin_amdgcn_wmma_f32_16x16x32_f16(false, a1[1][kt], false, bf1, (short)0, acc11, false, false);
        }
#pragma unroll
        for (int r2 = 0; r2 < 8; ++r2) {
            int m = r2 + hi * 8;
            hid[wv][0][m][col0] = (f16)silu_f(acc00[r2]);
            hid[wv][0][m][col1] = (f16)silu_f(acc01[r2]);
            hid[wv][1][m][col0] = (f16)silu_f(acc10[r2]);
            hid[wv][1][m][col1] = (f16)silu_f(acc11[r2]);
        }
    }

    // ---- A2 fragments from LDS (re-layout C->A via LDS round trip) ----
    v16h a2[2][KT2];
#pragma unroll
    for (int t = 0; t < 2; ++t) {
#pragma unroll
        for (int kt = 0; kt < KT2; ++kt) {
            v8h c0 = *(const v8h*)&hid[wv][t][ln][kt * 32 + hi * 8];
            v8h c1 = *(const v8h*)&hid[wv][t][ln][kt * 32 + 16 + hi * 8];
#pragma unroll
            for (int j = 0; j < 8; ++j) { a2[t][kt][j] = c0[j]; a2[t][kt][8 + j] = c1[j]; }
        }
    }

    int sidx[2][8], ridx[2][8];
#pragma unroll
    for (int t = 0; t < 2; ++t)
#pragma unroll
        for (int r2 = 0; r2 < 8; ++r2) {
            int m = edgeBase + t * 16 + r2 + hi * 8;
            sidx[t][r2] = send[m];
            ridx[t][r2] = recv[m];
        }
    const float* mrow = mult + (multPerBatch ? (size_t)b * NELEC * KER : 0);
    float* zb = z + (size_t)b * NELEC * KER;

    // ---- GEMM2 (B from global, double-buffered) + multiply + scatter ----
    for (int op = 0; op < NT2; op += 2) {
        v8f acc00 = {0.f,0.f,0.f,0.f,0.f,0.f,0.f,0.f};
        v8f acc01 = acc00, acc10 = acc00, acc11 = acc00;
        v16h bf0 = LDB(w2sw, NT2, 0, op);
        v16h bf1 = LDB(w2sw, NT2, 0, op + 1);
#pragma unroll
        for (int kt = 0; kt < KT2; ++kt) {
            v16h nb0 = bf0, nb1 = bf1;
            if (kt + 1 < KT2) {
                nb0 = LDB(w2sw, NT2, kt + 1, op);
                nb1 = LDB(w2sw, NT2, kt + 1, op + 1);
            }
            acc00 = __builtin_amdgcn_wmma_f32_16x16x32_f16(false, a2[0][kt], false, bf0, (short)0, acc00, false, false);
            acc10 = __builtin_amdgcn_wmma_f32_16x16x32_f16(false, a2[1][kt], false, bf0, (short)0, acc10, false, false);
            acc01 = __builtin_amdgcn_wmma_f32_16x16x32_f16(false, a2[0][kt], false, bf1, (short)0, acc01, false, false);
            acc11 = __builtin_amdgcn_wmma_f32_16x16x32_f16(false, a2[1][kt], false, bf1, (short)0, acc11, false, false);
            bf0 = nb0; bf1 = nb1;
        }
        int col0 = op * 16 + ln;
        int col1 = col0 + 16;
#pragma unroll
        for (int r2 = 0; r2 < 8; ++r2) {
            float v00 = acc00[r2] * mrow[(size_t)sidx[0][r2] * KER + col0];
            float v01 = acc01[r2] * mrow[(size_t)sidx[0][r2] * KER + col1];
            float v10 = acc10[r2] * mrow[(size_t)sidx[1][r2] * KER + col0];
            float v11 = acc11[r2] * mrow[(size_t)sidx[1][r2] * KER + col1];
            atomicAdd(zb + (size_t)ridx[0][r2] * KER + col0, v00);
            atomicAdd(zb + (size_t)ridx[0][r2] * KER + col1, v01);
            atomicAdd(zb + (size_t)ridx[1][r2] * KER + col0, v10);
            atomicAdd(zb + (size_t)ridx[1][r2] * KER + col1, v11);
        }
    }
}

// ---------------------------------------------------------------------------
// hx = elec @ h  (per batch, 128x128 @ 128x128), WMMA with on-the-fly f16 A.
// ---------------------------------------------------------------------------
__global__ __launch_bounds__(256, 1) void hx_kernel(
    const float* __restrict__ elec, const f16* __restrict__ hsw, float* __restrict__ hx)
{
    const int lane = threadIdx.x & 31, wv = threadIdx.x >> 5;
    const int ln = lane & 15, hi = lane >> 4;
    const int b = blockIdx.x;
    const float* arow = elec + ((size_t)b * NELEC + wv * 16 + ln) * EMB;
    v16h a[KTE];
#pragma unroll
    for (int kt = 0; kt < KTE; ++kt) {
#pragma unroll
        for (int j = 0; j < 8; ++j) {
            a[kt][j]     = (f16)arow[kt * 32 + hi * 8 + j];
            a[kt][8 + j] = (f16)arow[kt * 32 + 16 + hi * 8 + j];
        }
    }
    float* ob = hx + (size_t)b * NELEC * KER;
    for (int np = 0; np < NTE; np += 2) {
        v8f acc0 = {0.f,0.f,0.f,0.f,0.f,0.f,0.f,0.f};
        v8f acc1 = acc0;
#pragma unroll
        for (int kt = 0; kt < KTE; ++kt) {
            acc0 = __builtin_amdgcn_wmma_f32_16x16x32_f16(false, a[kt], false, LDB(hsw, NTE, kt, np),     (short)0, acc0, false, false);
            acc1 = __builtin_amdgcn_wmma_f32_16x16x32_f16(false, a[kt], false, LDB(hsw, NTE, kt, np + 1), (short)0, acc1, false, false);
        }
#pragma unroll
        for (int r2 = 0; r2 < 8; ++r2) {
            size_t row = (size_t)(wv * 16 + r2 + hi * 8) * KER;
            ob[row + np * 16 + ln]      = acc0[r2];
            ob[row + np * 16 + 16 + ln] = acc1[r2];
        }
    }
}

// ---------------------------------------------------------------------------
// elec += z_same@g_same + z_anti@g_anti + z_ne@g_ne (per batch), WMMA.
// z layout [3][NBATCH][128][128]; gsw layout [3][KTE*NTE*512] fragments.
// ---------------------------------------------------------------------------
__global__ __launch_bounds__(256, 1) void update_kernel(
    const float* __restrict__ z, const f16* __restrict__ gsw, float* __restrict__ elec)
{
    const int lane = threadIdx.x & 31, wv = threadIdx.x >> 5;
    const int ln = lane & 15, hi = lane >> 4;
    const int b = blockIdx.x;
    v16h a[3][KTE];
#pragma unroll
    for (int l = 0; l < 3; ++l) {
        const float* arow = z + (((size_t)l * NBATCH + b) * NELEC + wv * 16 + ln) * KER;
#pragma unroll
        for (int kt = 0; kt < KTE; ++kt) {
#pragma unroll
            for (int j = 0; j < 8; ++j) {
                a[l][kt][j]     = (f16)arow[kt * 32 + hi * 8 + j];
                a[l][kt][8 + j] = (f16)arow[kt * 32 + 16 + hi * 8 + j];
            }
        }
    }
    float* eb = elec + (size_t)b * NELEC * EMB;
    for (int np = 0; np < NTE; np += 2) {
        v8f acc0 = {0.f,0.f,0.f,0.f,0.f,0.f,0.f,0.f};
        v8f acc1 = acc0;
#pragma unroll
        for (int l = 0; l < 3; ++l) {
            const f16* gl = gsw + (size_t)l * (KTE * NTE * 512);
#pragma unroll
            for (int kt = 0; kt < KTE; ++kt) {
                acc0 = __builtin_amdgcn_wmma_f32_16x16x32_f16(false, a[l][kt], false, LDB(gl, NTE, kt, np),     (short)0, acc0, false, false);
                acc1 = __builtin_amdgcn_wmma_f32_16x16x32_f16(false, a[l][kt], false, LDB(gl, NTE, kt, np + 1), (short)0, acc1, false, false);
            }
        }
#pragma unroll
        for (int r2 = 0; r2 < 8; ++r2) {
            size_t row = (size_t)(wv * 16 + r2 + hi * 8) * EMB;
            eb[row + np * 16 + ln]      += acc0[r2];
            eb[row + np * 16 + 16 + ln] += acc1[r2];
        }
    }
}

// ---------------------------------------------------------------------------
// Host orchestration.
// Input order (jax pytree, dict keys sorted):
//   0 rs, 1 coords, 2 X, 3 Y,
//   per layer l (base = 4 + 13*l):
//     +0 g_anti +1 g_ne +2 g_same +3 h
//     +4 W1_anti +5 W2_anti +6 b1_anti
//     +7 W1_ne   +8 W2_ne   +9 b1_ne
//     +10 W1_same +11 W2_same +12 b1_same
//   43 s_same 44 r_same 45 s_anti 46 r_anti 47 s_ne 48 r_ne
// ---------------------------------------------------------------------------
extern "C" void kernel_launch(void* const* d_in, const int* in_sizes, int n_in,
                              void* d_out, int out_size, void* d_ws, size_t ws_size,
                              hipStream_t stream) {
    (void)in_sizes; (void)n_in; (void)out_size; (void)ws_size;
    const float* rs     = (const float*)d_in[0];
    const float* coords = (const float*)d_in[1];
    const float* X      = (const float*)d_in[2];
    const float* Y      = (const float*)d_in[3];
    const int* sS = (const int*)d_in[43]; const int* rS = (const int*)d_in[44];
    const int* sA = (const int*)d_in[45]; const int* rA = (const int*)d_in[46];
    const int* sN = (const int*)d_in[47]; const int* rN = (const int*)d_in[48];

    // internal label order: 0=same, 1=anti, 2=ne
    const float *W1p[3][3], *W2p[3][3], *B1p[3][3], *Hp[3], *Gp[3][3];
    for (int l = 0; l < 3; ++l) {
        int base = 4 + 13 * l;
        Gp[l][1]  = (const float*)d_in[base + 0];   // anti
        Gp[l][2]  = (const float*)d_in[base + 1];   // ne
        Gp[l][0]  = (const float*)d_in[base + 2];   // same
        Hp[l]     = (const float*)d_in[base + 3];
        W1p[l][1] = (const float*)d_in[base + 4];  W2p[l][1] = (const float*)d_in[base + 5];  B1p[l][1] = (const float*)d_in[base + 6];
        W1p[l][2] = (const float*)d_in[base + 7];  W2p[l][2] = (const float*)d_in[base + 8];  B1p[l][2] = (const float*)d_in[base + 9];
        W1p[l][0] = (const float*)d_in[base + 10]; W2p[l][0] = (const float*)d_in[base + 11]; B1p[l][0] = (const float*)d_in[base + 12];
    }

    // Workspace layout
    char* ws = (char*)d_ws;
    size_t off = 0;
    f16* featsS = (f16*)(ws + off); off += (size_t)NBATCH * E_SAME * INW * sizeof(f16);
    f16* featsA = (f16*)(ws + off); off += (size_t)NBATCH * E_ANTI * INW * sizeof(f16);
    f16* featsN = (f16*)(ws + off); off += (size_t)NBATCH * E_NE   * INW * sizeof(f16);
    float* hx = (float*)(ws + off); off += (size_t)NBATCH * NELEC * KER * sizeof(float);
    float* z  = (float*)(ws + off); off += (size_t)3 * NBATCH * NELEC * KER * sizeof(float);
    const size_t W1SZ = W1_HALFS;                   // halfs per fragment-matrix
    const size_t W2SZ = (size_t)KT2 * NT2 * 512;
    const size_t HSZ  = (size_t)KTE * NTE * 512;
    const size_t LHALFS = 3 * W1SZ + 3 * W2SZ + HSZ + 3 * HSZ;
    f16* wbase = (f16*)(ws + off);

    float* elec = (float*)d_out;    // elec lives in d_out through all layers

    // ---- weight conversion + fragment swizzle (f32 -> f16, padded) ----
    for (int l = 0; l < 3; ++l) {
        f16* w1sw = wbase + l * LHALFS;
        f16* w2sw = w1sw + 3 * W1SZ;
        f16* hsw  = w2sw + 3 * W2SZ;
        f16* gsw  = hsw + HSZ;
        for (int lab = 0; lab < 3; ++lab) {
            swizzle_w<<<(int)(W1SZ / 256), 256, 0, stream>>>(W1p[l][lab], w1sw + lab * W1SZ, NT1, INW, HID_REAL);
            swizzle_w<<<(int)(W2SZ / 256), 256, 0, stream>>>(W2p[l][lab], w2sw + lab * W2SZ, NT2, HID_REAL, KER);
            swizzle_w<<<(int)(HSZ  / 256), 256, 0, stream>>>(Gp[l][lab],  gsw  + lab * HSZ,  NTE, KER, EMB);
        }
        swizzle_w<<<(int)(HSZ / 256), 256, 0, stream>>>(Hp[l], hsw, NTE, EMB, KER);
    }

    // ---- geometric features (once, shared across layers) ----
    feat_kernel<<<dim3(E_SAME / 8, NBATCH), 256, 0, stream>>>(rs, coords, sS, rS, featsS, E_SAME, 0);
    feat_kernel<<<dim3(E_ANTI / 8, NBATCH), 256, 0, stream>>>(rs, coords, sA, rA, featsA, E_ANTI, 0);
    feat_kernel<<<dim3(E_NE   / 8, NBATCH), 256, 0, stream>>>(rs, coords, sN, rN, featsN, E_NE,   1);

    // ---- elec init: broadcast X ----
    init_elec<<<(NBATCH * NELEC * EMB) / 256, 256, 0, stream>>>(elec, X);

    const size_t ZL = (size_t)NBATCH * NELEC * KER;
    for (int l = 0; l < 3; ++l) {
        f16* w1sw = wbase + l * LHALFS;
        f16* w2sw = w1sw + 3 * W1SZ;
        f16* hsw  = w2sw + 3 * W2SZ;
        f16* gsw  = hsw + HSZ;

        hx_kernel<<<NBATCH, 256, 0, stream>>>(elec, hsw, hx);
        zero_f32<<<(int)((3 * ZL) / 256), 256, 0, stream>>>(z, (int)(3 * ZL));

        edge_mlp<<<dim3(E_SAME / 128, NBATCH), 128, W1_LDS_BYTES, stream>>>(
            featsS, w1sw + 0 * W1SZ, B1p[l][0], w2sw + 0 * W2SZ, hx, 1, sS, rS, z + 0 * ZL, E_SAME);
        edge_mlp<<<dim3(E_ANTI / 128, NBATCH), 128, W1_LDS_BYTES, stream>>>(
            featsA, w1sw + 1 * W1SZ, B1p[l][1], w2sw + 1 * W2SZ, hx, 1, sA, rA, z + 1 * ZL, E_ANTI);
        edge_mlp<<<dim3(E_NE / 128, NBATCH), 128, W1_LDS_BYTES, stream>>>(
            featsN, w1sw + 2 * W1SZ, B1p[l][2], w2sw + 2 * W2SZ, Y, 0, sN, rN, z + 2 * ZL, E_NE);

        update_kernel<<<NBATCH, 256, 0, stream>>>(z, gsw, elec);
    }
}
